// Head_38311108280672
// MI455X (gfx1250) — matrix operand order repaired and checked by reading, not compiled, not run on hardware
//
#include <hip/hip_runtime.h>

typedef __attribute__((ext_vector_type(16))) _Float16 v16h;
typedef __attribute__((ext_vector_type(8)))  _Float16 v8h;
typedef __attribute__((ext_vector_type(8)))  float    v8f;

// workspace layout (bytes)
#define WS_WT  0                          // 3 * 64*192 f16 transposed weights
#define WS_Q   73728                      // 32768*64 f16
#define WS_K   (73728 + 4194304)
#define WS_V   (73728 + 2*4194304)        // V transposed: [B][64][2048] f16

static __device__ __forceinline__ v16h cat8(v8h lo, v8h hi) {
  v16h r;
#pragma unroll
  for (int i = 0; i < 8; ++i) { r[i] = lo[i]; r[i + 8] = hi[i]; }
  return r;
}

// ---------------- kernel 0: weights f32 -> transposed f16 ----------------
__global__ void prep_weights(const float* __restrict__ Wq,
                             const float* __restrict__ Wk,
                             const float* __restrict__ Wv,
                             _Float16* __restrict__ wt) {
  int tid = blockIdx.x * 256 + threadIdx.x;           // 3*64*192 = 36864
  if (tid >= 3 * 64 * 192) return;
  int mat = tid / (64 * 192);
  int rem = tid % (64 * 192);
  int n = rem / 192;                                   // output col of W
  int c = rem % 192;
  const float* W = (mat == 0) ? Wq : (mat == 1 ? Wk : Wv);
  wt[(size_t)mat * 64 * 192 + n * 192 + c] = (_Float16)W[c * 64 + n];
}

// ---------------- kernel 1: QKV projection (WMMA f16) ----------------
__global__ __launch_bounds__(128) void qkv_kernel(const float* __restrict__ x,
                                                  const _Float16* __restrict__ wt,
                                                  _Float16* __restrict__ Q16,
                                                  _Float16* __restrict__ K16,
                                                  _Float16* __restrict__ V16T) {
  __shared__ _Float16 xs[16 * 192];                   // x tile as f16
  const int tid  = threadIdx.x;
  const int lane = tid & 31, wave = tid >> 5;
  const int half = lane >> 4, l16 = lane & 15;
  const size_t R0 = (size_t)blockIdx.x * 16;          // flat row base (b*2048+t)

  // cooperative load x tile -> LDS f16 (24 floats per thread)
  const float* xg = x + R0 * 192;
#pragma unroll
  for (int i = 0; i < 6; ++i) {
    int off = tid * 24 + i * 4;
    float4 f = *(const float4*)(xg + off);
    _Float16* d = xs + off;
    d[0] = (_Float16)f.x; d[1] = (_Float16)f.y;
    d[2] = (_Float16)f.z; d[3] = (_Float16)f.w;
  }
  __syncthreads();

  const int n0 = wave * 16;                            // column group of H
  v8f cq = {}, ck = {}, cv = {};
  const _Float16* xrow = xs + l16 * 192;
  const _Float16* wq = wt + (n0 + l16) * 192;
  const _Float16* wk = wq + 64 * 192;
  const _Float16* wv = wk + 64 * 192;
#pragma unroll
  for (int kb = 0; kb < 6; ++kb) {                     // K = 192 = 6 * 32
    v8h lo = *(const v8h*)(xrow + kb * 32 + half * 8);
    v8h hi = *(const v8h*)(xrow + kb * 32 + 16 + half * 8);
    v16h a = cat8(lo, hi);
    v16h bq = cat8(*(const v8h*)(wq + kb * 32 + half * 8), *(const v8h*)(wq + kb * 32 + 16 + half * 8));
    v16h bk = cat8(*(const v8h*)(wk + kb * 32 + half * 8), *(const v8h*)(wk + kb * 32 + 16 + half * 8));
    v16h bv = cat8(*(const v8h*)(wv + kb * 32 + half * 8), *(const v8h*)(wv + kb * 32 + 16 + half * 8));
    cq = __builtin_amdgcn_wmma_f32_16x16x32_f16(false, a, false, bq, (short)0, cq, false, false);
    ck = __builtin_amdgcn_wmma_f32_16x16x32_f16(false, a, false, bk, (short)0, ck, false, false);
    cv = __builtin_amdgcn_wmma_f32_16x16x32_f16(false, a, false, bv, (short)0, cv, false, false);
  }
  const size_t Rl = R0 + half * 8;
  const int b = (int)(R0 >> 11);
  const int t_base = (int)(R0 & 2047) + half * 8;
#pragma unroll
  for (int r = 0; r < 8; ++r) {
    size_t row = Rl + r;
    Q16[row * 64 + n0 + l16] = (_Float16)cq[r];
    K16[row * 64 + n0 + l16] = (_Float16)ck[r];
    V16T[(size_t)b * 64 * 2048 + (size_t)(n0 + l16) * 2048 + (t_base + r)] = (_Float16)cv[r];
  }
}

// ---------------- kernel 2: flash attention, 64-wide s-chunks ----------------
__global__ __launch_bounds__(256) void attn_kernel(const _Float16* __restrict__ Q16,
                                                   const _Float16* __restrict__ K16,
                                                   const _Float16* __restrict__ V16T,
                                                   float* __restrict__ out) {
  __shared__ _Float16 ps[8 * 16 * 64];                 // per-wave P staging (2KB/wave)
  const int tid  = threadIdx.x;
  const int lane = tid & 31, wave = tid >> 5;
  const int half = lane >> 4, l16 = lane & 15;
  const int j  = blockIdx.x * 8 + wave;                // q-tile id, 0..2047
  const int b  = j >> 7;
  const int t0 = (j & 127) * 16;
  _Float16* psw = ps + wave * (16 * 64);

  // Q A-fragments for both k-steps (H = 64 = 2*32), kept in registers
  const _Float16* qrow = Q16 + ((size_t)b * 2048 + t0 + l16) * 64;
  v16h aq0 = cat8(*(const v8h*)(qrow + half * 8),      *(const v8h*)(qrow + 16 + half * 8));
  v16h aq1 = cat8(*(const v8h*)(qrow + 32 + half * 8), *(const v8h*)(qrow + 48 + half * 8));

  v8f o[4];
  float m[8], l[8];
#pragma unroll
  for (int i = 0; i < 4; ++i) { v8f z = {}; o[i] = z; }
#pragma unroll
  for (int r = 0; r < 8; ++r) { m[r] = -1e30f; l[r] = 0.f; }

  const float sc_qk = 0.125f;                          // 64^-0.5
  const int nch = (t0 >> 6) + 1;                       // chunks of 64 cols
  const _Float16* kbase = K16  + (size_t)b * 2048 * 64;
  const _Float16* vbase = V16T + (size_t)b * 64 * 2048;

  for (int cu = 0; cu < nch; ++cu) {
    const int s0 = cu * 64;

    // prefetch next chunk's K rows and V columns into near caches
    if (cu + 1 < nch) {
      const int sn = s0 + 64;
      __builtin_prefetch(kbase + (size_t)(sn + lane) * 64, 0, 0);
      __builtin_prefetch(kbase + (size_t)(sn + 32 + lane) * 64, 0, 0);
      __builtin_prefetch(vbase + (size_t)lane * 2048 + sn, 0, 0);
      __builtin_prefetch(vbase + (size_t)(32 + lane) * 2048 + sn, 0, 0);
    }

    // S = Q K^T : four 16-wide col tiles, 2 k-steps each
    v8f st[4];
#pragma unroll
    for (int i = 0; i < 4; ++i) { v8f z = {}; st[i] = z; }
#pragma unroll
    for (int i = 0; i < 4; ++i) {
      const _Float16* kr = kbase + (size_t)(s0 + i * 16 + l16) * 64;
      v16h bk0 = cat8(*(const v8h*)(kr + half * 8),      *(const v8h*)(kr + 16 + half * 8));
      v16h bk1 = cat8(*(const v8h*)(kr + 32 + half * 8), *(const v8h*)(kr + 48 + half * 8));
      st[i] = __builtin_amdgcn_wmma_f32_16x16x32_f16(false, aq0, false, bk0, (short)0, st[i], false, false);
      st[i] = __builtin_amdgcn_wmma_f32_16x16x32_f16(false, aq1, false, bk1, (short)0, st[i], false, false);
    }

    // scale + causal mask (in place)
    const bool need_mask = (s0 + 63) > t0;
#pragma unroll
    for (int i = 0; i < 4; ++i) {
#pragma unroll
      for (int r = 0; r < 8; ++r) {
        float a = st[i][r] * sc_qk;
        if (need_mask && (s0 + i * 16 + l16 > t0 + half * 8 + r)) a = -1e30f;
        st[i][r] = a;
      }
    }

    // online softmax: row max across 64 cols
    float mn[8], sf[8];
#pragma unroll
    for (int r = 0; r < 8; ++r) {
      float v = fmaxf(fmaxf(st[0][r], st[1][r]), fmaxf(st[2][r], st[3][r]));
      v = fmaxf(v, __shfl_xor(v, 1, 32));
      v = fmaxf(v, __shfl_xor(v, 2, 32));
      v = fmaxf(v, __shfl_xor(v, 4, 32));
      v = fmaxf(v, __shfl_xor(v, 8, 32));
      mn[r] = fmaxf(m[r], v);
      sf[r] = __expf(m[r] - mn[r]);
      m[r]  = mn[r];
    }

    // exp, row sum, stage P to LDS, rescale O
#pragma unroll
    for (int r = 0; r < 8; ++r) {
      float s = 0.f;
#pragma unroll
      for (int i = 0; i < 4; ++i) {
        float e = __expf(st[i][r] - mn[r]);
        s += e;
        psw[(half * 8 + r) * 64 + i * 16 + l16] = (_Float16)e;
      }
      s += __shfl_xor(s, 1, 32);
      s += __shfl_xor(s, 2, 32);
      s += __shfl_xor(s, 4, 32);
      s += __shfl_xor(s, 8, 32);
      l[r] = l[r] * sf[r] + s;
#pragma unroll
      for (int i = 0; i < 4; ++i) o[i][r] *= sf[r];
    }
    asm volatile("s_wait_dscnt 0x0" ::: "memory");      // LDS writes -> reads, in-wave

    // P A-fragments (16x64 -> two 16x32 k-steps)
    const _Float16* prow = psw + l16 * 64;
    v16h ap0 = cat8(*(const v8h*)(prow + half * 8),      *(const v8h*)(prow + 16 + half * 8));
    v16h ap1 = cat8(*(const v8h*)(prow + 32 + half * 8), *(const v8h*)(prow + 48 + half * 8));

    // O += P * V  (V transposed -> contiguous B fragments)
#pragma unroll
    for (int ht = 0; ht < 4; ++ht) {
      const _Float16* vr = vbase + (size_t)(ht * 16 + l16) * 2048 + s0;
      v16h bv0 = cat8(*(const v8h*)(vr + half * 8),      *(const v8h*)(vr + 16 + half * 8));
      v16h bv1 = cat8(*(const v8h*)(vr + 32 + half * 8), *(const v8h*)(vr + 48 + half * 8));
      o[ht] = __builtin_amdgcn_wmma_f32_16x16x32_f16(false, ap0, false, bv0, (short)0, o[ht], false, false);
      o[ht] = __builtin_amdgcn_wmma_f32_16x16x32_f16(false, ap1, false, bv1, (short)0, o[ht], false, false);
    }
  }

  // normalize and store f32 output
  float* og = out + ((size_t)b * 2048 + t0 + half * 8) * 64 + l16;
#pragma unroll
  for (int r = 0; r < 8; ++r) {
    float inv = 1.0f / l[r];
#pragma unroll
    for (int ht = 0; ht < 4; ++ht)
      og[(size_t)r * 64 + ht * 16] = o[ht][r] * inv;
  }
}

extern "C" void kernel_launch(void* const* d_in, const int* in_sizes, int n_in,
                              void* d_out, int out_size, void* d_ws, size_t ws_size,
                              hipStream_t stream) {
  const float* x  = (const float*)d_in[0];
  const float* Wq = (const float*)d_in[1];
  const float* Wk = (const float*)d_in[2];
  const float* Wv = (const float*)d_in[3];
  float* out = (float*)d_out;
  char* ws = (char*)d_ws;
  _Float16* wt  = (_Float16*)(ws + WS_WT);
  _Float16* Q16 = (_Float16*)(ws + WS_Q);
  _Float16* K16 = (_Float16*)(ws + WS_K);
  _Float16* V16 = (_Float16*)(ws + WS_V);

  prep_weights<<<144, 256, 0, stream>>>(Wq, Wk, Wv, wt);
  qkv_kernel<<<2048, 128, 0, stream>>>(x, wt, Q16, K16, V16);
  attn_kernel<<<256, 256, 0, stream>>>(Q16, K16, V16, out);
}
